// HOPEBlock_68856915689572
// MI455X (gfx1250) — compile-verified, matrix operand order checked by reading, not run on hardware
//
#include <hip/hip_runtime.h>

typedef __attribute__((ext_vector_type(16))) _Float16 v16h;
typedef __attribute__((ext_vector_type(8)))  _Float16 v8h;
typedef __attribute__((ext_vector_type(8)))  float    v8f;

// Async global->LDS path via inline asm (CDNA5 GLOBAL_LOAD_ASYNC_TO_LDS_B128,
// tracked by ASYNCcnt). Set to 0 to fall back to plain vector copies.
#define USE_ASYNC_LDS 1

// ---------------------------------------------------------------------------
// WMMA fragment helpers (CDNA5 / gfx1250, wave32)
// A fragment (16x32 f16): lane l holds row M = l%16; element e maps to
//   K = (e/8)*16 + (l/16)*8 + (e%8)  -> two contiguous 16B LDS loads per lane.
// B fragment mirrors A with lane = column N (we keep all B-side operands in
// [n][k] / transposed layouts so fragment loads are contiguous ds_load_b128).
// C/D (16x16 f32): VGPR r, lane l -> row = r + 8*(l/16), col = l%16.
// ---------------------------------------------------------------------------
__device__ __forceinline__ v16h ld_frag(const _Float16* rowptr, int lane) {
  const int klo = (lane >> 4) << 3;
  v8h lo = *(const v8h*)(rowptr + klo);
  v8h hi = *(const v8h*)(rowptr + 16 + klo);
  return __builtin_shufflevector(lo, hi, 0, 1, 2, 3, 4, 5, 6, 7,
                                 8, 9, 10, 11, 12, 13, 14, 15);
}

__device__ __forceinline__ v8f wmma16(v16h a, v16h b, v8f c) {
  return __builtin_amdgcn_wmma_f32_16x16x32_f16(false, a, false, b,
                                                (short)0, c, false, false);
}

// 16B global -> LDS copy. Async form issues GLOBAL_LOAD_ASYNC_TO_LDS_B128:
// VDST = LDS byte address (low 32 bits of the generic shared pointer);
// VADDR = 64-bit global address (GV
// mode, saddr = off).
__device__ __forceinline__ void cp16(const _Float16* g, _Float16* l) {
#if USE_ASYNC_LDS
  unsigned int loff = (unsigned int)(unsigned long long)l;
  asm volatile("global_load_async_to_lds_b128 %0, %1, off"
               :
               : "v"(loff), "v"((unsigned long long)g)
               : "memory");
#else
  *(v8h*)l = *(const v8h*)g;
#endif
}

__device__ __forceinline__ void async_fence() {
#if USE_ASYNC_LDS
  asm volatile("s_wait_asynccnt 0x0" ::: "memory");
#endif
}

// ---------------------------------------------------------------------------
// fp32 -> fp16 transposing converter: out[n*K + k] = (f16) in[k*N + n]
// Tiled through LDS so both sides stay coalesced. grid.z = weight planes.
// ---------------------------------------------------------------------------
__global__ __launch_bounds__(256) void cvt_t(const float* __restrict__ in,
                                             _Float16* __restrict__ out,
                                             int K, int N) {
  __shared__ _Float16 tile[32][33];
  const size_t plane = (size_t)K * N;
  const float* ip = in + blockIdx.z * plane;
  _Float16* op = out + blockIdx.z * plane;
  const int kb = blockIdx.y * 32, nb = blockIdx.x * 32;
#pragma unroll
  for (int j = 0; j < 4; ++j)
    tile[threadIdx.y + 8 * j][threadIdx.x] =
        (_Float16)ip[(size_t)(kb + threadIdx.y + 8 * j) * N + nb + threadIdx.x];
  __syncthreads();
#pragma unroll
  for (int j = 0; j < 4; ++j)
    op[(size_t)(nb + threadIdx.y + 8 * j) * K + kb + threadIdx.x] =
        tile[threadIdx.x][threadIdx.y + 8 * j];
}

// ---------------------------------------------------------------------------
// LayerNorm over rows of 1024, f32 in -> f16 out
// ---------------------------------------------------------------------------
__global__ __launch_bounds__(256) void ln_to_f16(const float* __restrict__ x,
                                                 const float* __restrict__ g,
                                                 const float* __restrict__ bta,
                                                 _Float16* __restrict__ out) {
  const int row = blockIdx.x;
  const float* xr = x + (size_t)row * 1024;
  float s = 0.f, sq = 0.f;
  for (int i = threadIdx.x; i < 1024; i += 256) {
    float v = xr[i];
    s += v;
    sq += v * v;
  }
  __shared__ float sh[256], sh2[256];
  sh[threadIdx.x] = s;
  sh2[threadIdx.x] = sq;
  __syncthreads();
  for (int off = 128; off > 0; off >>= 1) {
    if (threadIdx.x < off) {
      sh[threadIdx.x] += sh[threadIdx.x + off];
      sh2[threadIdx.x] += sh2[threadIdx.x + off];
    }
    __syncthreads();
  }
  const float mean = sh[0] * (1.f / 1024.f);
  const float var = sh2[0] * (1.f / 1024.f) - mean * mean;
  const float rstd = rsqrtf(var + 1e-5f);
  for (int i = threadIdx.x; i < 1024; i += 256)
    out[(size_t)row * 1024 + i] =
        (_Float16)((xr[i] - mean) * rstd * g[i] + bta[i]);
}

// ---------------------------------------------------------------------------
// Tiled WMMA GEMM: C = A(MxK,f16 row-major) * Bt(NxK,f16 pre-transposed).
// 128x128 tile, K-step 32, 8 waves x (2x4) 16x16 tiles, register-staged
// global->LDS pipeline so global latency overlaps WMMA compute.
// EPI: 0 = Q: per-head layout         4 = K: per-head layout + [bh][d][s] copy
//      5 = V: [bh][d][s] copy only    1 = +residual -> f32
//      2 = +bias tanh-GELU -> f16     3 = +bias -> f16
// ---------------------------------------------------------------------------
template <int EPI>
__global__ __launch_bounds__(256) void gemm_wmma_kernel(
    const _Float16* __restrict__ A, const _Float16* __restrict__ Bt,
    _Float16* __restrict__ outh, _Float16* __restrict__ outT,
    float* __restrict__ outf, const float* __restrict__ bias,
    const float* __restrict__ resid, int M, int N, int K) {
  __shared__ _Float16 As[128 * 32];  // [m][k]
  __shared__ _Float16 Bs[128 * 32];  // [n][k]

  const int tid = threadIdx.x;
  const int lane = tid & 31;
  const int wave = tid >> 5;
  const int wr = wave >> 1;  // 0..3 -> 32-row block
  const int wc = wave & 1;   // 0..1 -> 64-col block
  const long bm = (long)blockIdx.y * 128;
  const long bn = (long)blockIdx.x * 128;

  // fixed (row, k-offset) pair per thread for the two 16B staging chunks
  int rowj[2], kkj[2];
#pragma unroll
  for (int j = 0; j < 2; ++j) {
    int c = tid + j * 256;
    rowj[j] = c >> 2;
    kkj[j] = (c & 3) * 8;
  }

  v8f acc[2][4];
#pragma unroll
  for (int i = 0; i < 2; ++i)
#pragma unroll
    for (int j = 0; j < 4; ++j) acc[i][j] = (v8f){};

  v8h ra[2], rb[2];
#pragma unroll
  for (int j = 0; j < 2; ++j) {
    ra[j] = *(const v8h*)(A + (bm + rowj[j]) * (size_t)K + kkj[j]);
    rb[j] = *(const v8h*)(Bt + (bn + rowj[j]) * (size_t)K + kkj[j]);
  }

  for (int k0 = 0; k0 < K; k0 += 32) {
    __syncthreads();  // previous tile's readers done
#pragma unroll
    for (int j = 0; j < 2; ++j) {
      *(v8h*)(As + rowj[j] * 32 + kkj[j]) = ra[j];
      *(v8h*)(Bs + rowj[j] * 32 + kkj[j]) = rb[j];
    }
    __syncthreads();
    if (k0 + 32 < K) {  // stage next slice while WMMAs run
#pragma unroll
      for (int j = 0; j < 2; ++j) {
        const _Float16* sa = A + (bm + rowj[j]) * (size_t)K + k0 + 32 + kkj[j];
        const _Float16* sb = Bt + (bn + rowj[j]) * (size_t)K + k0 + 32 + kkj[j];
        ra[j] = *(const v8h*)sa;
        rb[j] = *(const v8h*)sb;
        if (k0 + 64 < K) {
          __builtin_prefetch(sa + 32, 0, 1);  // global_prefetch_b8
          __builtin_prefetch(sb + 32, 0, 1);
        }
      }
    }
    v16h af0 = ld_frag(As + (wr * 32 + (lane & 15)) * 32, lane);
    v16h af1 = ld_frag(As + (wr * 32 + 16 + (lane & 15)) * 32, lane);
#pragma unroll
    for (int ct = 0; ct < 4; ++ct) {
      v16h bf = ld_frag(Bs + (wc * 64 + ct * 16 + (lane & 15)) * 32, lane);
      acc[0][ct] = wmma16(af0, bf, acc[0][ct]);
      acc[1][ct] = wmma16(af1, bf, acc[1][ct]);
    }
  }

  const int lr = 8 * (lane >> 4);
  const int lc = lane & 15;
#pragma unroll
  for (int rt = 0; rt < 2; ++rt)
#pragma unroll
    for (int ct = 0; ct < 4; ++ct)
#pragma unroll
      for (int r = 0; r < 8; ++r) {
        long row = bm + wr * 32 + rt * 16 + lr + r;
        long col = bn + wc * 64 + ct * 16 + lc;
        float v = acc[rt][ct][r];
        if constexpr (EPI == 0 || EPI == 4 || EPI == 5) {
          long b = row >> 12, s = row & 4095;
          long bh = b * 16 + (col >> 6), d = col & 63;
          if constexpr (EPI != 5) outh[(bh * 4096 + s) * 64 + d] = (_Float16)v;
          if constexpr (EPI != 0) outT[(bh * 64 + d) * 4096 + s] = (_Float16)v;
        } else if constexpr (EPI == 1) {
          outf[row * (long)N + col] = v + resid[row * (long)N + col];
        } else if constexpr (EPI == 2) {
          float t = v + bias[col];
          float gl = 0.5f * t *
                     (1.f + tanhf(0.7978845608f * (t + 0.044715f * t * t * t)));
          outh[row * (long)N + col] = (_Float16)gl;
        } else {
          outh[row * (long)N + col] = (_Float16)(v + bias[col]);
        }
      }
}

// ---------------------------------------------------------------------------
// Chunked linear attention with per-head 64x64 memory state.
// One workgroup per (b, head); 8 waves. State kept as f32 WMMA accumulators
// (2 tiles/wave) + f16 LDS mirror for the inter GEMM. kT/vT come from global
// pre-transposed copies, so every LDS fill is a plain 16B async copy and
// every fragment load is a contiguous ds_load_b128. WMMA pairs interleaved to
// cover the D->A/B hazard latency with independent work.
// ---------------------------------------------------------------------------
__global__ __launch_bounds__(256) void attn_kernel(
    const _Float16* __restrict__ Q, const _Float16* __restrict__ Kn,
    const _Float16* __restrict__ KT, const _Float16* __restrict__ VT,
    _Float16* __restrict__ yflat) {
  extern __shared__ _Float16 sm[];
  _Float16* qs = sm;            // 128x64  [s][d]
  _Float16* ks = qs + 8192;     // 128x64  [s][d]
  _Float16* kTs = ks + 8192;    // 64x128  [d][s]
  _Float16* vTs = kTs + 8192;   // 64x128  [d][s]
  _Float16* Mhs = vTs + 8192;   // 64x64 f16 mirror of M
  _Float16* scs = Mhs + 4096;   // 128x128 masked scores

  const int tid = threadIdx.x, lane = tid & 31, w = tid >> 5;
  const int bh = blockIdx.x, b = bh >> 4, h = bh & 15;
  const _Float16* gq = Q + (size_t)bh * 4096 * 64;
  const _Float16* gk = Kn + (size_t)bh * 4096 * 64;
  const _Float16* gkT = KT + (size_t)bh * 64 * 4096;
  const _Float16* gvT = VT + (size_t)bh * 64 * 4096;

  for (int i = tid; i < 4096; i += 256) Mhs[i] = (_Float16)0.f;
  v8f Macc[2];
  Macc[0] = (v8f){};
  Macc[1] = (v8f){};
  __syncthreads();

  const int lc = lane & 15, lr = 8 * (lane >> 4);

  for (int ch = 0; ch < 32; ++ch) {
    const int s0 = ch * 128;
// fill q, k, kT, vT : 16B chunks, 4 per thread per matrix
#pragma unroll
    for (int j = 0; j < 4; ++j) {
      int c = tid + j * 256;
      int row = c >> 3, kk = (c & 7) * 8;    // 128 x 64 mapping
      int row2 = c >> 4, cc = (c & 15) * 8;  // 64 x 128 mapping
      cp16(gq + (size_t)(s0 + row) * 64 + kk, qs + row * 64 + kk);
      cp16(gk + (size_t)(s0 + row) * 64 + kk, ks + row * 64 + kk);
      cp16(gkT + (size_t)row2 * 4096 + s0 + cc, kTs + row2 * 128 + cc);
      cp16(gvT + (size_t)row2 * 4096 + s0 + cc, vTs + row2 * 128 + cc);
    }
    async_fence();
    __syncthreads();

    v16h aq0 = ld_frag(qs + (w * 16 + lc) * 64 + 0, lane);
    v16h aq1 = ld_frag(qs + (w * 16 + lc) * 64 + 32, lane);

    // inter-chunk: y = q * M^T  (B[k][v] = Mh[v][k]); tile pairs interleaved
    v8f y[4];
#pragma unroll
    for (int nt = 0; nt < 4; nt += 2) {
      v16h b00 = ld_frag(Mhs + (nt * 16 + lc) * 64 + 0, lane);
      v16h b01 = ld_frag(Mhs + (nt * 16 + lc) * 64 + 32, lane);
      v16h b10 = ld_frag(Mhs + ((nt + 1) * 16 + lc) * 64 + 0, lane);
      v16h b11 = ld_frag(Mhs + ((nt + 1) * 16 + lc) * 64 + 32, lane);
      v8f t0 = (v8f){}, t1 = (v8f){};
      t0 = wmma16(aq0, b00, t0);
      t1 = wmma16(aq0, b10, t1);
      t0 = wmma16(aq1, b01, t0);
      t1 = wmma16(aq1, b11, t1);
      y[nt] = t0;
      y[nt + 1] = t1;
    }

// masked scores strip (own 16 rows x 128 cols); column-tile pairs interleaved
#pragma unroll
    for (int ct = 0; ct < 8; ct += 2) {
      v16h b00 = ld_frag(ks + (ct * 16 + lc) * 64 + 0, lane);
      v16h b01 = ld_frag(ks + (ct * 16 + lc) * 64 + 32, lane);
      v16h b10 = ld_frag(ks + ((ct + 1) * 16 + lc) * 64 + 0, lane);
      v16h b11 = ld_frag(ks + ((ct + 1) * 16 + lc) * 64 + 32, lane);
      v8f s0v = (v8f){}, s1v = (v8f){};
      s0v = wmma16(aq0, b00, s0v);
      s1v = wmma16(aq0, b10, s1v);
      s0v = wmma16(aq1, b01, s0v);
      s1v = wmma16(aq1, b11, s1v);
#pragma unroll
      for (int p = 0; p < 2; ++p) {
        const int col = (ct + p) * 16 + lc;
        const v8f sv = p ? s1v : s0v;
#pragma unroll
        for (int r = 0; r < 8; ++r) {
          int rr = w * 16 + lr + r;
          scs[rr * 128 + col] = (col <= rr) ? (_Float16)sv[r] : (_Float16)0.f;
        }
      }
    }
    __syncthreads();

// intra-chunk: y += scores * v  (B from vT, contiguous over c')
#pragma unroll
    for (int ksI = 0; ksI < 4; ++ksI) {
      v16h a = ld_frag(scs + (w * 16 + lc) * 128 + ksI * 32, lane);
#pragma unroll
      for (int nt = 0; nt < 4; ++nt) {
        v16h bb = ld_frag(vTs + (nt * 16 + lc) * 128 + ksI * 32, lane);
        y[nt] = wmma16(a, bb, y[nt]);
      }
    }

// write y back to flat (b*S+s, h*64+d) layout for the Wo GEMM
#pragma unroll
    for (int nt = 0; nt < 4; ++nt) {
      long col = (long)h * 64 + nt * 16 + lc;
#pragma unroll
      for (int r = 0; r < 8; ++r) {
        long row = (long)b * 4096 + s0 + w * 16 + lr + r;
        yflat[row * 1024 + col] = (_Float16)y[nt][r];
      }
    }

    // state update: M[v][k] += sum_c vT[v][c] * kT[k][c] (2 independent tiles)
    const int mr = (w >> 1) * 16;
#pragma unroll
    for (int ksI = 0; ksI < 4; ++ksI) {
      v16h a = ld_frag(vTs + (mr + lc) * 128 + ksI * 32, lane);
#pragma unroll
      for (int j = 0; j < 2; ++j) {
        v16h bb = ld_frag(kTs + (((w & 1) * 2 + j) * 16 + lc) * 128 + ksI * 32,
                          lane);
        Macc[j] = wmma16(a, bb, Macc[j]);
      }
    }
// refresh the f16 mirror for next chunk's inter GEMM
#pragma unroll
    for (int j = 0; j < 2; ++j) {
      int colb = ((w & 1) * 2 + j) * 16 + lc;
#pragma unroll
      for (int r = 0; r < 8; ++r)
        Mhs[(mr + lr + r) * 64 + colb] = (_Float16)Macc[j][r];
    }
    __syncthreads();
  }
}

// ---------------------------------------------------------------------------
// out = x1 + (float)h
// ---------------------------------------------------------------------------
__global__ void add_residual(const float* __restrict__ x1,
                             const _Float16* __restrict__ hh,
                             float* __restrict__ out, int n) {
  for (int i = blockIdx.x * blockDim.x + threadIdx.x; i < n;
       i += gridDim.x * blockDim.x)
    out[i] = x1[i] + (float)hh[i];
}

// ---------------------------------------------------------------------------
extern "C" void kernel_launch(void* const* d_in, const int* in_sizes, int n_in,
                              void* d_out, int out_size, void* d_ws,
                              size_t ws_size, hipStream_t stream) {
  (void)in_sizes; (void)n_in; (void)out_size; (void)ws_size;
  const float* x      = (const float*)d_in[0];
  const float* ln1_g  = (const float*)d_in[1];
  const float* ln1_b  = (const float*)d_in[2];
  const float* wq     = (const float*)d_in[3];
  const float* wk     = (const float*)d_in[4];
  const float* wv     = (const float*)d_in[5];
  const float* wo     = (const float*)d_in[6];
  const float* ln2_g  = (const float*)d_in[7];
  const float* ln2_b  = (const float*)d_in[8];
  const float* cms_w1 = (const float*)d_in[9];
  const float* cms_b1 = (const float*)d_in[10];
  const float* cms_w2 = (const float*)d_in[11];
  const float* cms_b2 = (const float*)d_in[12];
  float* out = (float*)d_out;

  const size_t R = 16384;  // B*S rows
  size_t off = 0;
  auto alloc = [&](size_t bytes) -> void* {
    void* p = (char*)d_ws + off;
    off += (bytes + 255) & ~(size_t)255;
    return p;
  };
  _Float16* wqt = (_Float16*)alloc(1024 * 1024 * 2);  // [n][k] transposed
  _Float16* wkt = (_Float16*)alloc(1024 * 1024 * 2);
  _Float16* wvt = (_Float16*)alloc(1024 * 1024 * 2);
  _Float16* wot = (_Float16*)alloc(1024 * 1024 * 2);
  _Float16* w1t = (_Float16*)alloc((size_t)3 * 1024 * 4096 * 2);
  _Float16* w2t = (_Float16*)alloc((size_t)3 * 4096 * 1024 * 2);
  _Float16* h1  = (_Float16*)alloc(R * 1024 * 2);
  _Float16* qh  = (_Float16*)alloc(R * 1024 * 2);  // [bh][s][d]
  _Float16* kh  = (_Float16*)alloc(R * 1024 * 2);  // [bh][s][d]
  _Float16* kT  = (_Float16*)alloc(R * 1024 * 2);  // [bh][d][s]
  _Float16* vT  = (_Float16*)alloc(R * 1024 * 2);  // [bh][d][s]
  _Float16* yfl = (_Float16*)alloc(R * 1024 * 2);
  float*    x1  = (float*)alloc(R * 1024 * 4);
  _Float16* hA  = (_Float16*)alloc(R * 1024 * 2);
  _Float16* hB  = (_Float16*)alloc(R * 1024 * 2);
  _Float16* tmp = (_Float16*)alloc(R * 4096 * 2);

  // weight conversion + transpose (once per launch; tiny vs 23.3 TB/s)
  dim3 tb(32, 8);
  cvt_t<<<dim3(32, 32, 1), tb, 0, stream>>>(wq, wqt, 1024, 1024);
  cvt_t<<<dim3(32, 32, 1), tb, 0, stream>>>(wk, wkt, 1024, 1024);
  cvt_t<<<dim3(32, 32, 1), tb, 0, stream>>>(wv, wvt, 1024, 1024);
  cvt_t<<<dim3(32, 32, 1), tb, 0, stream>>>(wo, wot, 1024, 1024);
  cvt_t<<<dim3(128, 32, 3), tb, 0, stream>>>(cms_w1, w1t, 1024, 4096);
  cvt_t<<<dim3(32, 128, 3), tb, 0, stream>>>(cms_w2, w2t, 4096, 1024);

  // LN1 -> h1 (f16)
  ln_to_f16<<<16384, 256, 0, stream>>>(x, ln1_g, ln1_b, h1);

  // Q/K/V projections; K also emits [bh][d][s] copy, V emits transposed only
  dim3 g1024(8, 128);
  gemm_wmma_kernel<0><<<g1024, 256, 0, stream>>>(
      h1, wqt, qh, nullptr, nullptr, nullptr, nullptr, 16384, 1024, 1024);
  gemm_wmma_kernel<4><<<g1024, 256, 0, stream>>>(
      h1, wkt, kh, kT, nullptr, nullptr, nullptr, 16384, 1024, 1024);
  gemm_wmma_kernel<5><<<g1024, 256, 0, stream>>>(
      h1, wvt, nullptr, vT, nullptr, nullptr, nullptr, 16384, 1024, 1024);

  // chunked linear attention (64 blocks = B*HEADS), 104 KB dynamic LDS
  attn_kernel<<<64, 256, (size_t)53248 * sizeof(_Float16), stream>>>(
      qh, kh, kT, vT, yfl);

  // y @ wo + x -> x1 (f32)
  gemm_wmma_kernel<1><<<g1024, 256, 0, stream>>>(
      yfl, wot, nullptr, nullptr, x1, nullptr, x, 16384, 1024, 1024);

  // LN2 -> hA (f16)
  ln_to_f16<<<16384, 256, 0, stream>>>(x1, ln2_g, ln2_b, hA);

  // CMS: 3 chained GELU MLPs
  dim3 g4096(32, 128);
  _Float16* src = hA;
  _Float16* dst = hB;
  for (int l = 0; l < 3; ++l) {
    gemm_wmma_kernel<2><<<g4096, 256, 0, stream>>>(
        src, w1t + (size_t)l * 1024 * 4096, tmp, nullptr, nullptr,
        cms_b1 + l * 4096, nullptr, 16384, 4096, 1024);
    gemm_wmma_kernel<3><<<g1024, 256, 0, stream>>>(
        tmp, w2t + (size_t)l * 4096 * 1024, dst, nullptr, nullptr,
        cms_b2 + l * 1024, nullptr, 16384, 1024, 4096);
    _Float16* sw = src; src = dst; dst = sw;
  }

  // out = x1 + h
  add_residual<<<8192, 256, 0, stream>>>(x1, src, out, 16777216);
}
